// Model_70609262346595
// MI455X (gfx1250) — matrix operand order repaired and checked by reading, not compiled, not run on hardware
//
#include <hip/hip_runtime.h>
#include <hip/hip_bf16.h>

typedef __bf16 bf16_t;
typedef bf16_t bf16x8  __attribute__((ext_vector_type(8)));
typedef bf16_t bf16x16 __attribute__((ext_vector_type(16)));
typedef float  f32x8   __attribute__((ext_vector_type(8)));

union BFrag { bf16x16 v16; bf16x8 v8[2]; };

#define EPSLN 1e-6f
#define Bsz  4
#define Cch  8
#define Tlen 500
#define ROWS (Bsz*Tlen)   // 2000
#define M1   128
#define M2   2048
#define M3   32768
#define Fdim 4096
#define Ddim 128
#define SPLIT_SEQ 8
#define SPLIT_GEO 64

// ---------------- block reduction helper (sum + sumsq) ----------------
__device__ inline void blk_reduce2(float& a, float& b, float* sA, float* sB,
                                   int tid, int nt) {
  sA[tid] = a; sB[tid] = b; __syncthreads();
  for (int s = nt >> 1; s > 0; s >>= 1) {
    if (tid < s) { sA[tid] += sA[tid + s]; sB[tid] += sB[tid + s]; }
    __syncthreads();
  }
  a = sA[0]; b = sB[0]; __syncthreads();
}

// ---------------- fp32 -> bf16 weight conversion ----------------
__global__ void k_cvt(const float* __restrict__ s, bf16_t* __restrict__ d, int n) {
  int i = blockIdx.x * blockDim.x + threadIdx.x;
  int stride = gridDim.x * blockDim.x;
  for (; i < n; i += stride) d[i] = (bf16_t)s[i];
}

// ---------------- diag(softmax(log_cov)) ----------------
__global__ void k_adj(const float* __restrict__ L, float* __restrict__ diag) {
  int c = threadIdx.x;
  if (c < Cch) {
    const float* r = L + c * Cch;
    float mx = r[0];
    for (int i = 1; i < Cch; i++) mx = fmaxf(mx, r[i]);
    float s = 0.f;
    for (int i = 0; i < Cch; i++) s += expf(r[i] - mx);
    diag[c] = expf(r[c] - mx) / s;
  }
}

// ---------------- TCN block 1: dwconv(d=1)+LN+relu + X@res_w1 ----------------
__global__ __launch_bounds__(128) void k_tcn1(
    const float* __restrict__ X, const float* __restrict__ w1,
    const float* __restrict__ b1, const float* __restrict__ ls,
    const float* __restrict__ lb, const float* __restrict__ rw,
    const float* __restrict__ rb, float* __restrict__ T1) {
  __shared__ float sA[128], sB[128];
  int row = blockIdx.x, b = row / Tlen, t = row % Tlen;
  int j = threadIdx.x, cin = j >> 4;
  const float* Xb = X + (size_t)b * Cch * Tlen;
  const float* xc = Xb + cin * Tlen;
  float xm = (t >= 1) ? xc[t - 1] : 0.f;
  float x0 = xc[t];
  float xp = (t + 1 < Tlen) ? xc[t + 1] : 0.f;
  float conv = w1[j] * xm + w1[M1 + j] * x0 + w1[2 * M1 + j] * xp + b1[j];
  float s1 = conv, s2 = conv * conv;
  blk_reduce2(s1, s2, sA, sB, j, 128);
  float mu = s1 / 128.f, inv = rsqrtf(s2 / 128.f - mu * mu + EPSLN);
  float y = fmaxf((conv - mu) * inv * ls[j] + lb[j], 0.f);
  float res = rb[j];
#pragma unroll
  for (int c = 0; c < Cch; c++) res += Xb[c * Tlen + t] * rw[c * M1 + j];
  T1[(size_t)row * M1 + j] = y + res;
}

// ---------------- TCN block 2 conv part: dwconv(d=2)+LN+relu ----------------
__global__ __launch_bounds__(256) void k_tcn2(
    const float* __restrict__ T1, const float* __restrict__ w2,
    const float* __restrict__ b2, const float* __restrict__ ls,
    const float* __restrict__ lb, float* __restrict__ Y2) {
  __shared__ float sA[256], sB[256];
  int row = blockIdx.x, t = row % Tlen, tid = threadIdx.x;
  const float* r0 = T1 + (size_t)row * M1;
  const float* rm = r0 - 2 * M1;
  const float* rp = r0 + 2 * M1;
  bool hm = (t >= 2), hp = (t + 2 < Tlen);
  float cv[8], s1 = 0.f, s2 = 0.f;
#pragma unroll
  for (int u = 0; u < 8; u++) {
    int jj = u * 256 + tid, ci = jj >> 4;
    float xm = hm ? rm[ci] : 0.f, x0 = r0[ci], xp = hp ? rp[ci] : 0.f;
    float v = w2[jj] * xm + w2[M2 + jj] * x0 + w2[2 * M2 + jj] * xp + b2[jj];
    cv[u] = v; s1 += v; s2 += v * v;
  }
  blk_reduce2(s1, s2, sA, sB, tid, 256);
  float mu = s1 / (float)M2, inv = rsqrtf(s2 / (float)M2 - mu * mu + EPSLN);
#pragma unroll
  for (int u = 0; u < 8; u++) {
    int jj = u * 256 + tid;
    Y2[(size_t)row * M2 + jj] = fmaxf((cv[u] - mu) * inv * ls[jj] + lb[jj], 0.f);
  }
}

// ---------------- TCN block 3 conv part: dwconv(d=4)+LN+relu ----------------
__global__ __launch_bounds__(1024) void k_tcn3(
    const float* __restrict__ Y2, const float* __restrict__ w3,
    const float* __restrict__ b3, const float* __restrict__ ls,
    const float* __restrict__ lb, float* __restrict__ Y3) {
  __shared__ float sA[1024], sB[1024];
  int row = blockIdx.x, t = row % Tlen, tid = threadIdx.x;
  const float* r0 = Y2 + (size_t)row * M2;
  const float* rm = r0 - 4 * M2;
  const float* rp = r0 + 4 * M2;
  bool hm = (t >= 4), hp = (t + 4 < Tlen);
  float cv[32], s1 = 0.f, s2 = 0.f;
#pragma unroll
  for (int u = 0; u < 32; u++) {
    int jj = u * 1024 + tid, ci = jj >> 4;
    float xm = hm ? rm[ci] : 0.f, x0 = r0[ci], xp = hp ? rp[ci] : 0.f;
    float v = w3[jj] * xm + w3[M3 + jj] * x0 + w3[2 * M3 + jj] * xp + b3[jj];
    cv[u] = v; s1 += v; s2 += v * v;
  }
  blk_reduce2(s1, s2, sA, sB, tid, 1024);
  float mu = s1 / (float)M3, inv = rsqrtf(s2 / (float)M3 - mu * mu + EPSLN);
#pragma unroll
  for (int u = 0; u < 32; u++) {
    int jj = u * 1024 + tid;
    Y3[(size_t)row * M3 + jj] = fmaxf((cv[u] - mu) * inv * ls[jj] + lb[jj], 0.f);
  }
}

// ---------------- WMMA GEMM with async-to-LDS B staging + double buffering ----
// C[M,N] (+)= A[M,K](f32->bf16 in flight) @ B[K,N](bf16) [+ bias]
// Tile 128x128x32, 256 threads = 8 waves, wave tile 64x32 = 4x2 WMMA(16x16x32).
// gridDim.z = split-K factor. mode 0: direct store (+bias, optional accum).
// mode 1: store raw partial at Cio + z*M*N (reduced later by k_kred).
__global__ __launch_bounds__(256) void k_gemm(
    const float* __restrict__ A, const bf16_t* __restrict__ Bw,
    const float* __restrict__ bias, float* __restrict__ Cio,
    int M, int N, int K, int mode, int accum) {
  __shared__ bf16_t As[2][128][40];   // 32 + 8 pad (80B stride)
  __shared__ bf16_t Bs[2][32][136];   // 128 + 8 pad (272B stride)
  int tid = threadIdx.x;
  int lane = tid & 31, wave = tid >> 5;
  int wm = wave >> 2, wn = wave & 3;
  int m0 = blockIdx.x * 128, n0 = blockIdx.y * 128;
  int hi = lane >> 4, ln16 = lane & 15;

  int Kc = K / gridDim.z;
  int kbeg = blockIdx.z * Kc, kend = kbeg + Kc;

  f32x8 acc[4][2];
#pragma unroll
  for (int i = 0; i < 4; i++)
#pragma unroll
    for (int j = 0; j < 2; j++)
#pragma unroll
      for (int r = 0; r < 8; r++) acc[i][j][r] = 0.f;

  // A staging (sync, converts f32->bf16 through VALU)
  int arow = tid >> 1, ahalf = tid & 1;
  int gm_a = m0 + arow; if (gm_a >= M) gm_a = M - 1;  // clamp; stores guarded
  const float* aptr = A + (size_t)gm_a * K + ahalf * 16;
  // B staging (async to LDS, GVS mode: sgpr base + per-lane voffset)
  int brow = tid >> 3, bchunk = tid & 7;
  unsigned long long sbase = (unsigned long long)(const void*)(Bw + n0);
  unsigned int bvoff0 =
      (unsigned int)(((size_t)brow * N + (size_t)bchunk * 16) * sizeof(bf16_t));

  auto stageA = [&](int k0, int buf) {
    const float4* s4 = (const float4*)(aptr + k0);
    float4 v0 = s4[0], v1 = s4[1], v2 = s4[2], v3 = s4[3];
    float tmp[16] = {v0.x, v0.y, v0.z, v0.w, v1.x, v1.y, v1.z, v1.w,
                     v2.x, v2.y, v2.z, v2.w, v3.x, v3.y, v3.z, v3.w};
    bf16x8 o0, o1;
#pragma unroll
    for (int i = 0; i < 8; i++) { o0[i] = (bf16_t)tmp[i]; o1[i] = (bf16_t)tmp[8 + i]; }
    *(bf16x8*)&As[buf][arow][ahalf * 16]     = o0;
    *(bf16x8*)&As[buf][arow][ahalf * 16 + 8] = o1;
  };
  auto stageB = [&](int k0, int buf) {
    unsigned int voff = bvoff0 + (unsigned int)((size_t)k0 * N * sizeof(bf16_t));
    // generic shared address: low 32 bits are the raw LDS byte offset
    unsigned int loff =
        (unsigned int)(unsigned long long)(void*)&Bs[buf][brow][bchunk * 16];
    asm volatile("global_load_async_to_lds_b128 %0, %1, %2"
                 :: "v"(loff), "v"(voff), "s"(sbase) : "memory");
    asm volatile("global_load_async_to_lds_b128 %0, %1, %2"
                 :: "v"(loff + 16u), "v"(voff + 16u), "s"(sbase) : "memory");
  };

  stageA(kbeg, 0);
  stageB(kbeg, 0);
  asm volatile("s_wait_asynccnt 0" ::: "memory");
  __syncthreads();

  int buf = 0;
  for (int k0 = kbeg; k0 < kend; k0 += 32) {
    if (k0 + 32 < kend) { stageA(k0 + 32, buf ^ 1); stageB(k0 + 32, buf ^ 1); }

    BFrag bfr[2];
#pragma unroll
    for (int nt = 0; nt < 2; nt++) {   // B: lane = N col, K rows hi*8 / 16+hi*8
#pragma unroll
      for (int s = 0; s < 8; s++) {
        bfr[nt].v16[s]     = Bs[buf][hi * 8 + s][wn * 32 + nt * 16 + ln16];
        bfr[nt].v16[8 + s] = Bs[buf][16 + hi * 8 + s][wn * 32 + nt * 16 + ln16];
      }
    }
#pragma unroll
    for (int mt = 0; mt < 4; mt++) {   // A: lane = M row, K chunks hi*8 / 16+hi*8
      int mrow = wm * 64 + mt * 16 + ln16;
      BFrag af;
      af.v8[0] = *(const bf16x8*)&As[buf][mrow][hi * 8];
      af.v8[1] = *(const bf16x8*)&As[buf][mrow][16 + hi * 8];
#pragma unroll
      for (int nt = 0; nt < 2; nt++) {
        acc[mt][nt] = __builtin_amdgcn_wmma_f32_16x16x32_bf16(
            false, af.v16, false, bfr[nt].v16, (short)0, acc[mt][nt],
            false, false);
      }
    }
    asm volatile("s_wait_asynccnt 0" ::: "memory");
    __syncthreads();
    buf ^= 1;
  }

  // epilogue: D layout VGPR r -> M = r + 8*hi, N = lane&15
  size_t pbase = (size_t)blockIdx.z * (size_t)M * (size_t)N;
#pragma unroll
  for (int mt = 0; mt < 4; mt++) {
#pragma unroll
    for (int nt = 0; nt < 2; nt++) {
      int gn = n0 + wn * 32 + nt * 16 + ln16;
      float bv = (mode == 0) ? bias[gn] : 0.f;
#pragma unroll
      for (int r = 0; r < 8; r++) {
        int gm = m0 + wm * 64 + mt * 16 + hi * 8 + r;
        if (gm < M) {
          size_t idx = (size_t)gm * N + gn;
          if (mode == 0) {
            float v = acc[mt][nt][r] + bv;
            if (accum) v += Cio[idx];
            Cio[idx] = v;
          } else {
            Cio[pbase + idx] = acc[mt][nt][r];
          }
        }
      }
    }
  }
}

// ---------------- split-K reduction: C = bias + sum_z P[z] ----------------
__global__ void k_kred(const float* __restrict__ P, const float* __restrict__ bias,
                       float* __restrict__ C, int M, int N, int S) {
  int o = blockIdx.x * blockDim.x + threadIdx.x;
  if (o >= M * N) return;
  int n = o % N;
  float a = bias[n];
  for (int z = 0; z < S; z++) a += P[(size_t)z * M * N + o];
  C[o] = a;
}

// ---------------- GNN stage: diag-scale + LN(4096) + relu, in place ----------------
__global__ __launch_bounds__(256) void k_gnn(
    float* __restrict__ Y3, const float* __restrict__ diag,
    const float* __restrict__ ls, const float* __restrict__ lb) {
  __shared__ float sA[256], sB[256];
  int blk = blockIdx.x, c = blk & 7, row = blk >> 3, tid = threadIdx.x;
  float d = diag[c];
  float* base = Y3 + (size_t)row * M3 + c * Fdim;
  float v[16], s1 = 0.f, s2 = 0.f;
#pragma unroll
  for (int u = 0; u < 16; u++) {
    float x = base[u * 256 + tid] * d;
    v[u] = x; s1 += x; s2 += x * x;
  }
  blk_reduce2(s1, s2, sA, sB, tid, 256);
  float mu = s1 / (float)Fdim, inv = rsqrtf(s2 / (float)Fdim - mu * mu + EPSLN);
#pragma unroll
  for (int u = 0; u < 16; u++) {
    int f = u * 256 + tid;
    base[f] = fmaxf((v[u] - mu) * inv * ls[f] + lb[f], 0.f);
  }
}

// ---------------- geo segment mean: (B,10,32768) ----------------
__global__ void k_geo(const float* __restrict__ FLT, float* __restrict__ G) {
  int o = blockIdx.x * blockDim.x + threadIdx.x;
  if (o >= Bsz * 10 * M3) return;
  int f = o & (M3 - 1);
  int bs = o >> 15;
  int s = bs % 10, b = bs / 10;
  const float* p = FLT + ((size_t)(b * Tlen + s * 50)) * M3 + f;
  float a = 0.f;
  for (int i = 0; i < 50; i++) a += p[(size_t)i * M3];
  G[o] = a * (1.f / 50.f);
}

// ---------------- seq segment mean: (B,10,128) ----------------
__global__ void k_seqmean(const float* __restrict__ SI, float* __restrict__ SM) {
  int o = blockIdx.x * blockDim.x + threadIdx.x;
  if (o >= Bsz * 10 * Ddim) return;
  int d = o & 127;
  int bs = o >> 7;
  int s = bs % 10, b = bs / 10;
  const float* p = SI + ((size_t)(b * Tlen + s * 50)) * Ddim + d;
  float a = 0.f;
  for (int i = 0; i < 50; i++) a += p[i * Ddim];
  SM[o] = a * (1.f / 50.f);
}

// ---------------- head: LIF scan + attention + FC + mean spike ----------------
__global__ __launch_bounds__(512) void k_head(
    const float* __restrict__ SM, const float* __restrict__ GO,
    const float* __restrict__ lifw, const float* __restrict__ lifb,
    const float* __restrict__ aw, const float* __restrict__ ab,
    const float* __restrict__ f1w, const float* __restrict__ f1b,
    const float* __restrict__ f2w, const float* __restrict__ f2b,
    float* __restrict__ out) {
  __shared__ float spk[Bsz][10][Ddim];
  __shared__ float attn[Bsz][10];
  __shared__ float feat[Bsz][Ddim];
  __shared__ float h1[Bsz][64];
  __shared__ float red[512];
  int tid = threadIdx.x;
  int b = tid >> 7, d = tid & 127;
  float mem = 0.f, localsum = 0.f;
  for (int s = 0; s < 10; s++) {
    const float* g = GO + (size_t)(b * 10 + s) * Ddim;
    float z = lifb[d];
    for (int k = 0; k < Ddim; k++) z += g[k] * lifw[k * Ddim + d];
    float dec = 1.f / (1.f + expf(-z));
    mem = mem * dec + SM[(size_t)(b * 10 + s) * Ddim + d];
    float sp = (mem - 0.5f > 0.f) ? 1.f : 0.f;
    mem -= 0.5f * sp;
    spk[b][s][d] = sp;
    localsum += sp;
  }
  __syncthreads();
  if (tid < Bsz * 10) {
    int bb = tid / 10, ss = tid % 10;
    float a = ab[0];
    for (int k = 0; k < Ddim; k++) a += spk[bb][ss][k] * aw[k];
    attn[bb][ss] = a;
  }
  __syncthreads();
  if (tid < Bsz) {
    float mx = attn[tid][0];
    for (int s = 1; s < 10; s++) mx = fmaxf(mx, attn[tid][s]);
    float sm = 0.f;
    for (int s = 0; s < 10; s++) { float e = expf(attn[tid][s] - mx); attn[tid][s] = e; sm += e; }
    float inv = 1.f / sm;
    for (int s = 0; s < 10; s++) attn[tid][s] *= inv;
  }
  __syncthreads();
  { float fs = 0.f;
    for (int s = 0; s < 10; s++) fs += spk[b][s][d] * attn[b][s];
    feat[b][d] = fs; }
  __syncthreads();
  if (tid < Bsz * 64) {
    int bb = tid >> 6, h = tid & 63;
    float a = f1b[h];
    for (int k = 0; k < Ddim; k++) a += feat[bb][k] * f1w[k * 64 + h];
    h1[bb][h] = fmaxf(a, 0.f);
  }
  __syncthreads();
  if (tid < 16) {
    int bb = tid >> 2, o = tid & 3;
    float a = f2b[o];
    for (int k = 0; k < 64; k++) a += h1[bb][k] * f2w[k * 4 + o];
    out[bb * 4 + o] = a;
  }
  red[tid] = localsum; __syncthreads();
  for (int s = 256; s > 0; s >>= 1) {
    if (tid < s) red[tid] += red[tid + s];
    __syncthreads();
  }
  if (tid == 0) out[16] = red[0] / (float)(Bsz * 10 * Ddim);
}

// ---------------- host-side launch ----------------
extern "C" void kernel_launch(void* const* d_in, const int* in_sizes, int n_in,
                              void* d_out, int out_size, void* d_ws, size_t ws_size,
                              hipStream_t stream) {
  const float* X    = (const float*)d_in[0];
  const float* Lcov = (const float*)d_in[2];
  const float* cw1  = (const float*)d_in[3];
  const float* cb1  = (const float*)d_in[4];
  const float* l1s  = (const float*)d_in[5];
  const float* l1b  = (const float*)d_in[6];
  const float* rw1  = (const float*)d_in[7];
  const float* rb1  = (const float*)d_in[8];
  const float* cw2  = (const float*)d_in[9];
  const float* cb2  = (const float*)d_in[10];
  const float* l2s  = (const float*)d_in[11];
  const float* l2b  = (const float*)d_in[12];
  const float* rw2  = (const float*)d_in[13];
  const float* rb2  = (const float*)d_in[14];
  const float* cw3  = (const float*)d_in[15];
  const float* cb3  = (const float*)d_in[16];
  const float* l3s  = (const float*)d_in[17];
  const float* l3b  = (const float*)d_in[18];
  const float* rw3  = (const float*)d_in[19];
  const float* rb3  = (const float*)d_in[20];
  const float* gls  = (const float*)d_in[21];
  const float* glb  = (const float*)d_in[22];
  const float* seqw = (const float*)d_in[23];
  const float* seqb = (const float*)d_in[24];
  const float* geow = (const float*)d_in[25];
  const float* geob = (const float*)d_in[26];
  const float* lifw = (const float*)d_in[27];
  const float* lifb = (const float*)d_in[28];
  const float* aw   = (const float*)d_in[29];
  const float* ab   = (const float*)d_in[30];
  const float* f1w  = (const float*)d_in[31];
  const float* f1b  = (const float*)d_in[32];
  const float* f2w  = (const float*)d_in[33];
  const float* f2b  = (const float*)d_in[34];

  char* p = (char*)d_ws;
  auto carve = [&](size_t bytes) -> char* {
    char* r = p; p += (bytes + 255) & ~(size_t)255; return r;
  };
  bf16_t* W2B  = (bf16_t*)carve((size_t)M1 * M2 * 2);
  bf16_t* W3B  = (bf16_t*)carve((size_t)M2 * M3 * 2);
  bf16_t* SWB  = (bf16_t*)carve((size_t)M3 * Ddim * 2);
  bf16_t* GWB  = (bf16_t*)carve((size_t)M3 * Ddim * 2);
  float*  T1   = (float*)carve((size_t)ROWS * M1 * 4);
  float*  Y2   = (float*)carve((size_t)ROWS * M2 * 4);
  float*  Y3   = (float*)carve((size_t)ROWS * M3 * 4);   // also holds flat (in place)
  float*  GEOS = (float*)carve((size_t)Bsz * 10 * M3 * 4);
  float*  SEQI = (float*)carve((size_t)ROWS * Ddim * 4);
  float*  SEQM = (float*)carve((size_t)Bsz * 10 * Ddim * 4);
  float*  GEOO = (float*)carve((size_t)Bsz * 10 * Ddim * 4);
  float*  PSEQ = (float*)carve((size_t)SPLIT_SEQ * ROWS * Ddim * 4);
  float*  PGEO = (float*)carve((size_t)SPLIT_GEO * Bsz * 10 * Ddim * 4);
  float*  ADJ  = (float*)carve(64);

  // weight precision conversion (one pass; bf16 res_w3 fits L2 for GEMM reuse)
  k_cvt<<<512, 256, 0, stream>>>(rw2, W2B, M1 * M2);
  k_cvt<<<4096, 256, 0, stream>>>(rw3, W3B, M2 * M3);
  k_cvt<<<512, 256, 0, stream>>>(seqw, SWB, M3 * Ddim);
  k_cvt<<<512, 256, 0, stream>>>(geow, GWB, M3 * Ddim);
  k_adj<<<1, 32, 0, stream>>>(Lcov, ADJ);

  k_tcn1<<<ROWS, 128, 0, stream>>>(X, cw1, cb1, l1s, l1b, rw1, rb1, T1);
  k_tcn2<<<ROWS, 256, 0, stream>>>(T1, cw2, cb2, l2s, l2b, Y2);
  k_gemm<<<dim3(16, 16, 1), 256, 0, stream>>>(T1, W2B, rb2, Y2,
                                              ROWS, M2, M1, 0, 1);
  k_tcn3<<<ROWS, 1024, 0, stream>>>(Y2, cw3, cb3, l3s, l3b, Y3);
  k_gemm<<<dim3(16, 256, 1), 256, 0, stream>>>(Y2, W3B, rb3, Y3,
                                               ROWS, M3, M2, 0, 1);
  k_gnn<<<ROWS * Cch, 256, 0, stream>>>(Y3, ADJ, gls, glb);
  k_geo<<<(Bsz * 10 * M3 + 255) / 256, 256, 0, stream>>>(Y3, GEOS);
  // split-K tail GEMMs (deterministic partials, no float atomics)
  k_gemm<<<dim3(16, 1, SPLIT_SEQ), 256, 0, stream>>>(Y3, SWB, seqb, PSEQ,
                                                     ROWS, Ddim, M3, 1, 0);
  k_kred<<<(ROWS * Ddim + 255) / 256, 256, 0, stream>>>(PSEQ, seqb, SEQI,
                                                        ROWS, Ddim, SPLIT_SEQ);
  k_gemm<<<dim3(1, 1, SPLIT_GEO), 256, 0, stream>>>(GEOS, GWB, geob, PGEO,
                                                    Bsz * 10, Ddim, M3, 1, 0);
  k_kred<<<(Bsz * 10 * Ddim + 255) / 256, 256, 0, stream>>>(PGEO, geob, GEOO,
                                                            Bsz * 10, Ddim,
                                                            SPLIT_GEO);
  k_seqmean<<<(Bsz * 10 * Ddim + 255) / 256, 256, 0, stream>>>(SEQI, SEQM);
  k_head<<<1, 512, 0, stream>>>(SEQM, GEOO, lifw, lifb, aw, ab, f1w, f1b,
                                f2w, f2b, (float*)d_out);
}
